// TDNNKalmanNet_3238405341653
// MI455X (gfx1250) — matrix operand order repaired and checked by reading, not compiled, not run on hardware
//
#include <hip/hip_runtime.h>
#include <math.h>

// ---------------------------------------------------------------------------
// TDNN + Kalman + rollout-GRU for MI455X (gfx1250, wave32, WMMA 16x16x32 f16)
// ---------------------------------------------------------------------------

#define NB 64
#define NQ 512
#define ND 128
#define NH 256
#define NK3 3
#define WOUT 64
#define PI_F 3.14159265358979f

typedef __attribute__((ext_vector_type(16))) _Float16 v16h;
typedef __attribute__((ext_vector_type(8)))  float    v8f;

union frag16 { v16h v; unsigned u[8]; _Float16 h[16]; };

// A-matrix fragment (16x32 f16): lane L holds row M = L&15, g = L>>4.
// VGPR j<4 : K = 2j + 8g ;  j>=4 : K = 16 + 2(j-4) + 8g   (pairs as b32)
__device__ __forceinline__ v16h load_a_frag(const _Float16* row, int g) {
  frag16 f;
#pragma unroll
  for (int j = 0; j < 8; ++j) {
    int k = (j < 4) ? (2 * j + 8 * g) : (16 + 2 * (j - 4) + 8 * g);
    f.u[j] = *(const unsigned*)(row + k);
  }
  return f.v;
}

// B-matrix fragment (32x16 f16), stored transposed as [n][k] (k contiguous):
// lane L: col N = L&15, g = L>>4.  VGPR j<4 : K = 2j + 8g ;  j>=4 : K = 16 + 2(j-4) + 8g  (pairs as b32)
__device__ __forceinline__ v16h load_b_frag(const _Float16* row, int g) {
  frag16 f;
#pragma unroll
  for (int j = 0; j < 8; ++j)
    f.u[j] = *(const unsigned*)(row + ((j < 4) ? (2 * j + 8 * g) : (16 + 2 * (j - 4) + 8 * g)));
  return f.v;
}

__device__ __forceinline__ float sigmoidf_(float x) { return 1.0f / (1.0f + __expf(-x)); }
__device__ __forceinline__ float geluf_(float x)    { return 0.5f * x * (1.0f + erff(x * 0.70710678118f)); }

// ---------------------------------------------------------------------------
// weight conversion helpers
// ---------------------------------------------------------------------------
__global__ void k_cvt(const float* __restrict__ s, _Float16* __restrict__ d, int n) {
  int i = blockIdx.x * 256 + threadIdx.x;
  if (i < n) d[i] = (_Float16)s[i];
}

// combined Kalman weight: rows 0-1 = fc_rp_w, rows 2-129 = fc_gain_w, 130-143 = 0
__global__ void k_buildwk(const float* __restrict__ rpw, const float* __restrict__ rpb,
                          const float* __restrict__ gw,  const float* __restrict__ gb,
                          _Float16* __restrict__ wk, float* __restrict__ bk) {
  int i = blockIdx.x * 256 + threadIdx.x;
  if (i < 144 * NH) {
    int n = i >> 8, c = i & 255;
    float v = (n < 2) ? rpw[n * NH + c] : ((n < 130) ? gw[(n - 2) * NH + c] : 0.0f);
    wk[i] = (_Float16)v;
  }
  if (i < 144) bk[i] = (i < 2) ? rpb[i] : ((i < 130) ? gb[i - 2] : 0.0f);
}

// ---------------------------------------------------------------------------
// feats = concat([x, dy, ddy]) stored channel-major (B, 384, Q) as f16
// ---------------------------------------------------------------------------
__global__ void k_feats(const float* __restrict__ x, _Float16* __restrict__ feats) {
  size_t i = (size_t)blockIdx.x * 256 + threadIdx.x;
  if (i >= (size_t)NB * 384 * NQ) return;
  int q = (int)(i % NQ);
  size_t t = i / NQ;
  int c = (int)(t % 384);
  int b = (int)(t / 384);
  const float* xb = x + (size_t)b * NQ * ND;
  int d = c & 127, kind = c >> 7;
  float x0 = xb[(size_t)q * ND + d];
  float v;
  if (kind == 0)      v = x0;
  else if (kind == 1) v = (q >= 1) ? (x0 - xb[(size_t)(q - 1) * ND + d]) : 0.0f;
  else {
    if (q == 0)      v = 0.0f;
    else if (q == 1) v = x0 - xb[d];
    else             v = x0 - 2.0f * xb[(size_t)(q - 1) * ND + d] + xb[(size_t)(q - 2) * ND + d];
  }
  feats[i] = (_Float16)v;
}

// ---------------------------------------------------------------------------
// Conv1d (edge-padded, dilated) as WMMA GEMM.
//   x   : (B, Cin, Q)  (float or f16)
//   Wh  : (256, Cin, ks) f16
//   out : (B, 256, Q) f32,  out = [gelu](W*x + bias)
// Block = 256 thr (8 waves); tile = 64 cout x 32 q; wave tile = 16x16.
// Per K-chunk we stage ONE q-window (32 + 2*pad rows) + all ks weight taps,
// then issue ks WMMAs -> 3x fewer barriers/LDS rounds than per-tap staging.
// ---------------------------------------------------------------------------
template <typename Tin, bool GELU>
__global__ __launch_bounds__(256) void k_conv(const Tin* __restrict__ x,
                                              const _Float16* __restrict__ Wh,
                                              const float* __restrict__ bias,
                                              float* __restrict__ out,
                                              int Cin, int ks, int dil) {
  __shared__ __align__(16) _Float16 xs[36][32];      // [q-window row][c]
  __shared__ __align__(16) _Float16 wsm[3][64][32];  // [tap][o][c]
  int b  = blockIdx.z;
  int q0 = blockIdx.x * 32;
  int o0 = blockIdx.y * 64;
  int tid = threadIdx.x, lane = tid & 31, wave = tid >> 5;
  int wq = wave & 1, wc = wave >> 1;
  int g = lane >> 4, l15 = lane & 15;

  int pad   = ((ks - 1) / 2) * dil;   // 0, 1 or 2
  int nrows = 32 + 2 * pad;           // <= 36

  v8f acc = {};
  const Tin* xb = x + (size_t)b * Cin * NQ;

  for (int c0 = 0; c0 < Cin; c0 += 32) {
    __syncthreads();
    { // stage x q-window (coalesced on q), f32/f16 -> f16, edge clamp
      int qv = tid & 31;
      int cb = tid >> 5;  // 0..7
      for (int rbase = 0; rbase < nrows; rbase += 32) {
        int r = rbase + qv;
        if (r < nrows) {
          int qq = q0 + r - pad;
          qq = qq < 0 ? 0 : (qq > NQ - 1 ? NQ - 1 : qq);
#pragma unroll
          for (int i = 0; i < 4; ++i) {
            int c = cb + 8 * i;
            xs[r][c] = (_Float16)(float)xb[(size_t)(c0 + c) * NQ + qq];
          }
        }
      }
      if (c0 + 32 < Cin)  // hint next K-chunk into cache
        __builtin_prefetch(&xb[(size_t)(c0 + 32 + cb) * NQ + q0 + qv], 0, 1);
    }
    { // stage weight tile, all taps of this chunk
      int o  = tid >> 2;
      int cb = (tid & 3) * 8;
#pragma unroll
      for (int i = 0; i < 8; ++i) {
        int c = cb + i;
        size_t base = ((size_t)(o0 + o) * Cin + (c0 + c)) * ks;
        for (int tap = 0; tap < ks; ++tap)
          wsm[tap][o][c] = Wh[base + tap];
      }
    }
    __syncthreads();
    for (int tap = 0; tap < ks; ++tap) {
      v16h af = load_a_frag(&wsm[tap][wc * 16 + l15][0], g);
      v16h bf = load_b_frag(&xs[wq * 16 + l15 + tap * dil][0], g);
      acc = __builtin_amdgcn_wmma_f32_16x16x32_f16(false, af, false, bf,
                                                   (short)0, acc, false, false);
    }
  }
#pragma unroll
  for (int r = 0; r < 8; ++r) {
    int o = wc * 16 + r + 8 * g;
    int q = wq * 16 + l15;
    float v = acc[r] + bias[o0 + o];
    if (GELU) v = geluf_(v);
    out[((size_t)b * NH + (o0 + o)) * NQ + q0 + q] = v;
  }
}

// ---------------------------------------------------------------------------
// residual + LayerNorm over channels: h = LN(h + y)   (h,y: (B,256,Q))
// one wave per (b,q)
// ---------------------------------------------------------------------------
__global__ __launch_bounds__(256) void k_resln(float* __restrict__ h, const float* __restrict__ y,
                                               const float* __restrict__ lw, const float* __restrict__ lb) {
  int wave = threadIdx.x >> 5, lane = threadIdx.x & 31;
  int idx = blockIdx.x * 8 + wave;
  int b = idx >> 9, q = idx & 511;
  size_t base = (size_t)b * NH * NQ + q;
  float v[8], s = 0.0f;
#pragma unroll
  for (int i = 0; i < 8; ++i) {
    int c = lane + 32 * i;
    v[i] = h[base + (size_t)c * NQ] + y[base + (size_t)c * NQ];
    s += v[i];
  }
  for (int m = 16; m; m >>= 1) s += __shfl_xor(s, m, 32);
  float mean = s * (1.0f / NH), vs = 0.0f;
#pragma unroll
  for (int i = 0; i < 8; ++i) { float d = v[i] - mean; vs += d * d; }
  for (int m = 16; m; m >>= 1) vs += __shfl_xor(vs, m, 32);
  float rstd = rsqrtf(vs * (1.0f / NH) + 1e-5f);
#pragma unroll
  for (int i = 0; i < 8; ++i) {
    int c = lane + 32 * i;
    h[base + (size_t)c * NQ] = (v[i] - mean) * rstd * lw[c] + lb[c];
  }
}

// out LayerNorm: h (B,256,Q) -> h_seq (B,Q,256) f32 + f16 copy
__global__ __launch_bounds__(256) void k_outln(const float* __restrict__ h,
                                               const float* __restrict__ lw, const float* __restrict__ lb,
                                               float* __restrict__ hs, _Float16* __restrict__ hsh) {
  int wave = threadIdx.x >> 5, lane = threadIdx.x & 31;
  int idx = blockIdx.x * 8 + wave;
  int b = idx >> 9, q = idx & 511;
  size_t base = (size_t)b * NH * NQ + q;
  float v[8], s = 0.0f;
#pragma unroll
  for (int i = 0; i < 8; ++i) { int c = lane + 32 * i; v[i] = h[base + (size_t)c * NQ]; s += v[i]; }
  for (int m = 16; m; m >>= 1) s += __shfl_xor(s, m, 32);
  float mean = s * (1.0f / NH), vs = 0.0f;
#pragma unroll
  for (int i = 0; i < 8; ++i) { float d = v[i] - mean; vs += d * d; }
  for (int m = 16; m; m >>= 1) vs += __shfl_xor(vs, m, 32);
  float rstd = rsqrtf(vs * (1.0f / NH) + 1e-5f);
  size_t ro = (size_t)idx * NH;
#pragma unroll
  for (int i = 0; i < 8; ++i) {
    int c = lane + 32 * i;
    float nv = (v[i] - mean) * rstd * lw[c] + lb[c];
    hs[ro + c] = nv;
    hsh[ro + c] = (_Float16)nv;
  }
}

// ---------------------------------------------------------------------------
// Kalman precompute: rpg = h_seq @ Wk^T + bk   (M=B*Q, N=144, K=256)
// direct global fragment loads (K contiguous in both operands)
// ---------------------------------------------------------------------------
__global__ __launch_bounds__(256) void k_rpg(const _Float16* __restrict__ A,
                                             const _Float16* __restrict__ Wk,
                                             const float* __restrict__ bk,
                                             float* __restrict__ rpg) {
  int wave = threadIdx.x >> 5, lane = threadIdx.x & 31;
  int tile = blockIdx.x * 8 + wave; // 2048 m-tiles x 9 n-tiles
  int tm = tile / 9, tn = tile % 9;
  int g = lane >> 4, l15 = lane & 15;
  const _Float16* Ar = A  + (size_t)(tm * 16 + l15) * NH;
  const _Float16* Br = Wk + (size_t)(tn * 16 + l15) * NH;
  v8f acc = {};
  for (int kc = 0; kc < NH; kc += 32) {
    v16h af = load_a_frag(Ar + kc, g);
    v16h bf = load_b_frag(Br + kc, g);
    acc = __builtin_amdgcn_wmma_f32_16x16x32_f16(false, af, false, bf,
                                                 (short)0, acc, false, false);
  }
#pragma unroll
  for (int r = 0; r < 8; ++r) {
    int row = tm * 16 + r + 8 * g;
    int col = tn * 16 + l15;
    rpg[(size_t)row * 144 + col] = acc[r] + bk[col];
  }
}

// ---------------------------------------------------------------------------
// Kalman scan: 512 sequential steps, state in registers; 1 block / batch
// ---------------------------------------------------------------------------
__global__ __launch_bounds__(64) void k_kalman(const float* __restrict__ x_in,
                                               const float* __restrict__ rpg,
                                               float* __restrict__ xpost) {
  int b = blockIdx.x, t = threadIdx.x; // t < 64
  const float* xb = x_in + (size_t)b * NQ * ND;
  float re = xb[t], im = xb[64 + t];
  for (int q = 0; q < NQ; ++q) {
    const float* rp = rpg + ((size_t)b * NQ + q) * 144;
    float rho = 1.25f * sigmoidf_(rp[0]);
    float phi = PI_F * tanhf(rp[1]);
    float c = __cosf(phi), s = __sinf(phi);
    float pr = rho * (c * re - s * im);
    float pj = rho * (s * re + c * im);
    float kr = sigmoidf_(rp[2 + t]);
    float ki = sigmoidf_(rp[2 + 64 + t]);
    float yr = xb[(size_t)q * ND + t];
    float yi = xb[(size_t)q * ND + 64 + t];
    re = pr + kr * (yr - pr);
    im = pj + ki * (yi - pj);
  }
  xpost[b * ND + t] = re;
  xpost[b * ND + 64 + t] = im;
}

// ---------------------------------------------------------------------------
// Rollout: one persistent 512-thread workgroup, 64 sequential GRU steps.
// State in LDS (~240 KB): zx(64x384 f16, cols0-255 = h_r f16), z(64x256 f16),
// hr(64x256 f32), ht(64x256 f32), cur(64x128 f32). All GEMMs via WMMA.
// ---------------------------------------------------------------------------
__global__ __launch_bounds__(512) void k_rollout(
    const float* __restrict__ hseq_f, const float* __restrict__ xpost,
    const _Float16* __restrict__ Wroll, const float* __restrict__ broll,
    const _Float16* __restrict__ Wih, const _Float16* __restrict__ Whh,
    const float* __restrict__ bih, const float* __restrict__ bhh,
    const float* __restrict__ lnw, const float* __restrict__ lnb,
    const float* __restrict__ Wrp, const float* __restrict__ brp,
    float* __restrict__ out) {
  extern __shared__ __align__(16) char sm[];
  _Float16* zx = (_Float16*)sm;            // 64 x 384
  _Float16* zb = zx + 64 * 384;            // 64 x 256
  float* hr  = (float*)(zb + 64 * 256);    // 64 x 256
  float* ht  = hr + 64 * 256;              // 64 x 256
  float* cur = ht + 64 * 256;              // 64 x 128

  int tid = threadIdx.x, lane = tid & 31, wave = tid >> 5; // 16 waves
  int g = lane >> 4, l15 = lane & 15;

  // init: h_r = h_seq[:, Q-1, :], cur = x_post
  for (int i = tid; i < 64 * 256; i += 512) {
    int bb = i >> 8, c = i & 255;
    float v = hseq_f[((size_t)bb * NQ + (NQ - 1)) * NH + c];
    hr[i] = v;
    zx[bb * 384 + c] = (_Float16)v;
  }
  for (int i = tid; i < 64 * 128; i += 512) cur[i] = xpost[i];
  __syncthreads();

  for (int step = 0; step < WOUT; ++step) {
    // P1: zx cols 256..383 = f16(cur)
    for (int i = tid; i < 64 * 128; i += 512) {
      int bb = i >> 7, c = i & 127;
      zx[bb * 384 + 256 + c] = (_Float16)cur[i];
    }
    __syncthreads();

    // P2: z = tanh(zx @ Wroll^T + broll)  (M=64,N=256,K=384) -> 64 tiles
#pragma unroll
    for (int tt = 0; tt < 4; ++tt) {
      int tile = wave * 4 + tt;
      int tm = tile >> 4, tn = tile & 15;
      const _Float16* Ar = zx + (tm * 16 + l15) * 384;
      const _Float16* Br = Wroll + (size_t)(tn * 16 + l15) * 384;
      v8f acc = {};
      for (int kc = 0; kc < 384; kc += 32) {
        v16h af = load_a_frag(Ar + kc, g);
        v16h bf = load_b_frag(Br + kc, g);
        acc = __builtin_amdgcn_wmma_f32_16x16x32_f16(false, af, false, bf,
                                                     (short)0, acc, false, false);
      }
#pragma unroll
      for (int r = 0; r < 8; ++r) {
        int bb = tm * 16 + r + 8 * g;
        int c = tn * 16 + l15;
        zb[bb * 256 + c] = (_Float16)tanhf(acc[r] + broll[c]);
      }
    }
    __syncthreads();

    // P3: GRU gates. Per 16x16 tile: ir+hr and iz+hz chain through one
    // accumulator each; inn and hn kept separate (n = tanh(inn + r*hn)).
#pragma unroll
    for (int tt = 0; tt < 4; ++tt) {
      int tile = wave * 4 + tt;
      int tm = tile >> 4, tn = tile & 15;
      const _Float16* Az = zb + (tm * 16 + l15) * 256;
      const _Float16* Ah = zx + (tm * 16 + l15) * 384; // cols 0..255 = h_r f16
      const _Float16* Bir = Wih + (size_t)(tn * 16 + l15) * 256;
      const _Float16* Bhr = Whh + (size_t)(tn * 16 + l15) * 256;
      const _Float16* Biz = Wih + (size_t)(256 + tn * 16 + l15) * 256;
      const _Float16* Bhz = Whh + (size_t)(256 + tn * 16 + l15) * 256;
      const _Float16* Bin = Wih + (size_t)(512 + tn * 16 + l15) * 256;
      const _Float16* Bhn = Whh + (size_t)(512 + tn * 16 + l15) * 256;
      v8f ar = {}, az = {}, ai = {}, ah = {};
      for (int kc = 0; kc < 256; kc += 32) {
        v16h faz = load_a_frag(Az + kc, g);
        v16h fah = load_a_frag(Ah + kc, g);
        ar = __builtin_amdgcn_wmma_f32_16x16x32_f16(false, faz, false, load_b_frag(Bir + kc, g), (short)0, ar, false, false);
        ar = __builtin_amdgcn_wmma_f32_16x16x32_f16(false, fah, false, load_b_frag(Bhr + kc, g), (short)0, ar, false, false);
        az = __builtin_amdgcn_wmma_f32_16x16x32_f16(false, faz, false, load_b_frag(Biz + kc, g), (short)0, az, false, false);
        az = __builtin_amdgcn_wmma_f32_16x16x32_f16(false, fah, false, load_b_frag(Bhz + kc, g), (short)0, az, false, false);
        ai = __builtin_amdgcn_wmma_f32_16x16x32_f16(false, faz, false, load_b_frag(Bin + kc, g), (short)0, ai, false, false);
        ah = __builtin_amdgcn_wmma_f32_16x16x32_f16(false, fah, false, load_b_frag(Bhn + kc, g), (short)0, ah, false, false);
      }
#pragma unroll
      for (int r = 0; r < 8; ++r) {
        int bb = tm * 16 + r + 8 * g;
        int c = tn * 16 + l15;
        float hh = hr[bb * 256 + c];
        float rg = sigmoidf_(ar[r] + bih[c] + bhh[c]);
        float zg = sigmoidf_(az[r] + bih[256 + c] + bhh[256 + c]);
        float nn = tanhf(ai[r] + bih[512 + c] + rg * (ah[r] + bhh[512 + c]));
        ht[bb * 256 + c] = (1.0f - zg) * nn + zg * hh;
      }
    }
    __syncthreads();

    // P4: per-batch LN(ht) -> hr (+f16 into zx), rp head, rotate cur, emit pred
#pragma unroll
    for (int j = 0; j < 4; ++j) {
      int row = wave + 16 * j;
      float v[8], s = 0.0f;
#pragma unroll
      for (int i = 0; i < 8; ++i) { v[i] = ht[row * 256 + lane + 32 * i]; s += v[i]; }
      for (int m = 16; m; m >>= 1) s += __shfl_xor(s, m, 32);
      float mean = s * (1.0f / NH), vs = 0.0f;
#pragma unroll
      for (int i = 0; i < 8; ++i) { float d = v[i] - mean; vs += d * d; }
      for (int m = 16; m; m >>= 1) vs += __shfl_xor(vs, m, 32);
      float rstd = rsqrtf(vs * (1.0f / NH) + 1e-5f);
      float d0 = 0.0f, d1 = 0.0f;
#pragma unroll
      for (int i = 0; i < 8; ++i) {
        int c = lane + 32 * i;
        float nv = (v[i] - mean) * rstd * lnw[c] + lnb[c];
        hr[row * 256 + c] = nv;
        zx[row * 384 + c] = (_Float16)nv;
        d0 += nv * Wrp[c];
        d1 += nv * Wrp[NH + c];
      }
      for (int m = 16; m; m >>= 1) { d0 += __shfl_xor(d0, m, 32); d1 += __shfl_xor(d1, m, 32); }
      d0 = __shfl(d0, 0, 32);
      d1 = __shfl(d1, 0, 32);
      float rho = 1.25f * sigmoidf_(d0 + brp[0]);
      float phi = PI_F * tanhf(d1 + brp[1]);
      float cc = __cosf(phi), ss = __sinf(phi);
#pragma unroll
      for (int p = 0; p < 2; ++p) {
        int i = lane + 32 * p;
        float re = cur[row * 128 + i], im = cur[row * 128 + 64 + i];
        float nr = rho * (cc * re - ss * im);
        float ni = rho * (ss * re + cc * im);
        cur[row * 128 + i] = nr;
        cur[row * 128 + 64 + i] = ni;
        out[((size_t)row * WOUT + step) * ND + i] = nr;
        out[((size_t)row * WOUT + step) * ND + 64 + i] = ni;
      }
    }
    __syncthreads();
  }
}

// ---------------------------------------------------------------------------
extern "C" void kernel_launch(void* const* d_in, const int* in_sizes, int n_in,
                              void* d_out, int out_size, void* d_ws, size_t ws_size,
                              hipStream_t stream) {
  (void)in_sizes; (void)n_in; (void)out_size; (void)ws_size;
  const float* x_in   = (const float*)d_in[0];
  const float* inp_w  = (const float*)d_in[2];
  const float* inp_b  = (const float*)d_in[3];
  const float* cw[4]  = { (const float*)d_in[4], (const float*)d_in[6],
                          (const float*)d_in[10], (const float*)d_in[12] };
  const float* cb[4]  = { (const float*)d_in[5], (const float*)d_in[7],
                          (const float*)d_in[11], (const float*)d_in[13] };
  const float* blnw[2] = { (const float*)d_in[8],  (const float*)d_in[14] };
  const float* blnb[2] = { (const float*)d_in[9],  (const float*)d_in[15] };
  const float* out_ln_w = (const float*)d_in[16];
  const float* out_ln_b = (const float*)d_in[17];
  const float* fc_gain_w = (const float*)d_in[18];
  const float* fc_gain_b = (const float*)d_in[19];
  const float* fc_rp_w = (const float*)d_in[20];
  const float* fc_rp_b = (const float*)d_in[21];
  const float* roll_in_w = (const float*)d_in[22];
  const float* roll_in_b = (const float*)d_in[23];
  const float* gru_w_ih = (const float*)d_in[24];
  const float* gru_w_hh = (const float*)d_in[25];
  const float* gru_b_ih = (const float*)d_in[26];
  const float* gru_b_hh = (const float*)d_in[27];
  const float* roll_ln_w = (const float*)d_in[28];
  const float* roll_ln_b = (const float*)d_in[29];
  const float* fc_rp_r_w = (const float*)d_in[30];
  const float* fc_rp_r_b = (const float*)d_in[31];
  float* out = (float*)d_out;

  // -------- workspace carve --------
  char* p = (char*)d_ws;
  auto take = [&](size_t bytes) -> char* {
    char* r = p; p += (bytes + 255) & ~(size_t)255; return r;
  };
  const size_t BHQ4 = (size_t)NB * NH * NQ * 4;
  float*    y1f   = (float*)take(BHQ4);         // aliases feats (feats dead before y1 written)
  _Float16* feats = (_Float16*)y1f;             // (B,384,Q) f16 = 24 MB < 32 MB
  float*    y2f   = (float*)take(BHQ4);
  float*    hf    = (float*)take(BHQ4);
  float*    hseqf = (float*)take((size_t)NB * NQ * NH * 4);
  _Float16* hseqh = (_Float16*)take((size_t)NB * NQ * NH * 2);
  float*    rpg   = (float*)take((size_t)NB * NQ * 144 * 4);
  float*    xpost = (float*)take((size_t)NB * ND * 4);
  _Float16* wInp  = (_Float16*)take((size_t)NH * 384 * 2);
  _Float16* wch[4];
  for (int i = 0; i < 4; ++i) wch[i] = (_Float16*)take((size_t)NH * NH * NK3 * 2);
  _Float16* wkh   = (_Float16*)take((size_t)144 * NH * 2);
  float*    wkb   = (float*)take(144 * 4);
  _Float16* wroll = (_Float16*)take((size_t)NH * 384 * 2);
  _Float16* wih   = (_Float16*)take((size_t)3 * NH * NH * 2);
  _Float16* whh   = (_Float16*)take((size_t)3 * NH * NH * 2);

  // -------- weight conversions --------
  auto cvt = [&](const float* s, _Float16* d, int n) {
    k_cvt<<<(n + 255) / 256, 256, 0, stream>>>(s, d, n);
  };
  cvt(inp_w, wInp, NH * 384);
  for (int i = 0; i < 4; ++i) cvt(cw[i], wch[i], NH * NH * NK3);
  cvt(roll_in_w, wroll, NH * 384);
  cvt(gru_w_ih, wih, 3 * NH * NH);
  cvt(gru_w_hh, whh, 3 * NH * NH);
  k_buildwk<<<144, 256, 0, stream>>>(fc_rp_w, fc_rp_b, fc_gain_w, fc_gain_b, wkh, wkb);

  // -------- feature stack (WMMA convs) --------
  {
    size_t n = (size_t)NB * 384 * NQ;
    k_feats<<<(unsigned)((n + 255) / 256), 256, 0, stream>>>(x_in, feats);
  }
  dim3 cgrid(NQ / 32, NH / 64, NB);
  k_conv<_Float16, false><<<cgrid, 256, 0, stream>>>(feats, wInp, inp_b, hf, 384, 1, 1);
  for (int blk = 0; blk < 2; ++blk) {
    int dil = 1 << blk;
    k_conv<float, true><<<cgrid, 256, 0, stream>>>(hf,  wch[blk * 2 + 0], cb[blk * 2 + 0], y1f, NH, NK3, dil);
    k_conv<float, true><<<cgrid, 256, 0, stream>>>(y1f, wch[blk * 2 + 1], cb[blk * 2 + 1], y2f, NH, NK3, dil);
    k_resln<<<NB * NQ / 8, 256, 0, stream>>>(hf, y2f, blnw[blk], blnb[blk]);
  }
  k_outln<<<NB * NQ / 8, 256, 0, stream>>>(hf, out_ln_w, out_ln_b, hseqf, hseqh);

  // -------- Kalman: batched head GEMM + register-resident scan --------
  k_rpg<<<(NB * NQ / 16) * 9 / 8, 256, 0, stream>>>(hseqh, wkh, wkb, rpg);
  k_kalman<<<NB, 64, 0, stream>>>(x_in, rpg, xpost);

  // -------- rollout GRU: single persistent WGP, ~240 KB LDS --------
  size_t roll_lds = (size_t)(64 * 384 + 64 * 256) * 2 + (size_t)(64 * 256 * 2 + 64 * 128) * 4;
  k_rollout<<<1, 512, roll_lds, stream>>>(hseqf, xpost, wroll, roll_in_b,
                                          wih, whh, gru_b_ih, gru_b_hh,
                                          roll_ln_w, roll_ln_b,
                                          fc_rp_r_w, fc_rp_r_b, out);
}
